// GraphConvolution_36807869727114
// MI455X (gfx1250) — matrix operand order repaired and checked by reading, not compiled, not run on hardware
//
#include <hip/hip_runtime.h>
#include <hip/hip_bf16.h>

// ---------------------------------------------------------------------------
// GCN layer: out = relu( adj @ (x @ W^T + b) )
//   x   [16384, 256]  f32
//   adj [16384,16384] f32   <- 1 GiB, pure stream from HBM (the bottleneck)
//   W   [64, 256]     f32
//   b   [64]          f32
//   out [16384, 64]   f32
//
// Strategy:
//   K1: hiddenT[64][16384] = (x @ W^T + b)^T, stored bf16 in workspace (2 MB,
//       L2-resident on MI455X's 192 MB L2).
//   K2: adj @ hidden via v_wmma_f32_16x16x32_bf16, adj converted f32->bf16 in
//       registers (v_cvt_pk_bf16_f32), fp32 accumulation, ReLU on store.
//       adj is streamed with NT hints so it cannot evict hiddenT from L2;
//       WGP-scope prefetch runs 16 K-steps ahead of the load clause.
//       Roofline: 1 GiB / 23.3 TB/s ~= 46 us; bf16 WMMA has >5x compute
//       headroom at that bandwidth, so the kernel stays memory-bound.
// ---------------------------------------------------------------------------

typedef __attribute__((ext_vector_type(16))) __bf16 v16bf;
typedef __attribute__((ext_vector_type(8)))  float  v8f;
typedef __attribute__((ext_vector_type(4)))  float  v4f;   // native vector for NT loads

#define GN   16384   // rows of x / adj
#define INF  256
#define OUTF 64

// ---------------- Kernel 1: hiddenT (bf16) = (x W^T + b)^T ----------------
__global__ void gcn_linear_bias_T(const float* __restrict__ x,
                                  const float* __restrict__ W,
                                  const float* __restrict__ bias,
                                  __bf16* __restrict__ hiddenT) {
    __shared__ float xs[4][INF];
    const int tid     = threadIdx.x;          // 256 threads
    const int rowBase = blockIdx.x * 4;       // 4 rows of x per block

    // cooperative load of 4 x-rows (1024 floats) into LDS
    #pragma unroll
    for (int i = 0; i < 4; ++i) {
        int idx = tid + i * 256;
        int r = idx >> 8, c = idx & 255;
        xs[r][c] = x[(size_t)(rowBase + r) * INF + c];
    }
    __syncthreads();

    const int o = tid & 63;   // output feature
    const int r = tid >> 6;   // row within block

    const v4f* wrow = (const v4f*)(W + (size_t)o * INF);
    const v4f* xrow = (const v4f*)(xs[r]);

    float acc = 0.0f;
    #pragma unroll 8
    for (int kk = 0; kk < INF / 4; ++kk) {
        v4f wv = wrow[kk];
        v4f xv = xrow[kk];
        acc += wv.x * xv.x + wv.y * xv.y + wv.z * xv.z + wv.w * xv.w;
    }
    acc += bias[o];
    hiddenT[(size_t)o * GN + rowBase + r] = (__bf16)acc;
}

// ---------------- Kernel 2: out = relu(adj @ hidden) via WMMA bf16 --------
// Block: 256 threads = 8 waves. M_TILE = 32 rows, N = 64 cols.
// Wave w: mt = w>>2 (two 16-row tiles), nt = w&3 (four 16-col tiles).

// One K=32 WMMA step: stream 16 fp32 of adj (NT), cvt to bf16, load the
// 32-byte bf16 B strip, and accumulate.
__device__ __forceinline__ void wmma_step(const float* __restrict__ aRow,
                                          int colBase,
                                          const __bf16* __restrict__ bCol,
                                          int k, v8f& c) {
    // A fragment: 16 fp32 -> 16 bf16 (non-temporal: adj has zero reuse)
    v4f a0 = __builtin_nontemporal_load((const v4f*)(aRow + colBase));
    v4f a1 = __builtin_nontemporal_load((const v4f*)(aRow + colBase + 4));
    v4f a2 = __builtin_nontemporal_load((const v4f*)(aRow + colBase + 8));
    v4f a3 = __builtin_nontemporal_load((const v4f*)(aRow + colBase + 12));

    v16bf va;
    va[0]  = (__bf16)a0.x; va[1]  = (__bf16)a0.y;
    va[2]  = (__bf16)a0.z; va[3]  = (__bf16)a0.w;
    va[4]  = (__bf16)a1.x; va[5]  = (__bf16)a1.y;
    va[6]  = (__bf16)a1.z; va[7]  = (__bf16)a1.w;
    va[8]  = (__bf16)a2.x; va[9]  = (__bf16)a2.y;
    va[10] = (__bf16)a2.z; va[11] = (__bf16)a2.w;
    va[12] = (__bf16)a3.x; va[13] = (__bf16)a3.y;
    va[14] = (__bf16)a3.z; va[15] = (__bf16)a3.w;

    // B fragment: 16 contiguous bf16 (32 bytes, 32B-aligned, L2-hot)
    v16bf vb = *(const v16bf*)(bCol + k);

    // D = A x B + C   (8 args: neg_a, A, neg_b, B, c_mod, C, reuse_a, reuse_b)
    c = __builtin_amdgcn_wmma_f32_16x16x32_bf16(
            false, va, false, vb, (short)0, c, false, false);
}

__global__ void __launch_bounds__(256)
gcn_aggregate_wmma(const float* __restrict__ adj,
                   const __bf16* __restrict__ hiddenT,
                   float* __restrict__ out) {
    const int tid  = threadIdx.x;
    const int lane = tid & 31;
    const int wave = tid >> 5;       // 0..7 (wave32)
    const int mt   = wave >> 2;      // 0..1
    const int nt   = wave & 3;       // 0..3
    const int hlf  = lane >> 4;      // 0: lanes 0-15, 1: lanes 16-31
    const int lm   = lane & 15;

    // A-operand row for this lane (16-bit A 16x32 layout):
    //   lanes 0-15 hold K = k..k+15 (this code's contiguous order, same as B) of row M=lm
    //   lanes 16-31 hold K = k+16..k+31 of row M=lm
    const int aRowIdx = blockIdx.x * 32 + mt * 16 + lm;
    const float* __restrict__ aRow = adj + (size_t)aRowIdx * GN;

    // B-operand column for this lane (bf16 B 32x16 layout):
    //   lane holds a contiguous 16-element K strip of column N=lm;
    //   lanes 0-15 -> K = k..k+15, lanes 16-31 -> K = k+16..k+31
    const int nCol = nt * 16 + lm;
    const __bf16* __restrict__ bCol = hiddenT + (size_t)nCol * GN + hlf * 16;

    v8f c = {};  // fp32 accumulator, 16x16 tile

    const int PF = 512;  // prefetch distance in elements (16 K-steps, 2 KB)
    int k = 0;

    // Main loop: unconditional WGP-scope prefetch (locality 3 -> near cache)
    #pragma unroll 2
    for (; k < GN - PF; k += 32) {
        __builtin_prefetch(aRow + k + hlf * 8 + PF, 0, 3);
        wmma_step(aRow, k + hlf * 16, bCol, k, c);
    }
    // Tail: last 16 K-steps, no prefetch
    #pragma unroll 2
    for (; k < GN; k += 32) {
        wmma_step(aRow, k + hlf * 16, bCol, k, c);
    }

    // Epilogue: c[r] -> row (base + r + 8*hlf), col (nt*16 + lm); ReLU.
    // Output is write-once -> non-temporal stores.
    const int outRowBase = blockIdx.x * 32 + mt * 16 + hlf * 8;
    float* __restrict__ o = out + (size_t)outRowBase * OUTF + nt * 16 + lm;
    #pragma unroll
    for (int r = 0; r < 8; ++r) {
        float v = c[r];
        v = v > 0.0f ? v : 0.0f;
        __builtin_nontemporal_store(v, o + (size_t)r * OUTF);
    }
}

// ---------------------------------------------------------------------------
extern "C" void kernel_launch(void* const* d_in, const int* in_sizes, int n_in,
                              void* d_out, int out_size, void* d_ws, size_t ws_size,
                              hipStream_t stream) {
    const float* x    = (const float*)d_in[0];
    const float* adj  = (const float*)d_in[1];
    const float* W    = (const float*)d_in[2];
    const float* bias = (const float*)d_in[3];
    float*       out  = (float*)d_out;
    __bf16*      hT   = (__bf16*)d_ws;   // 64*16384*2 = 2 MB

    gcn_linear_bias_T<<<GN / 4, 256, 0, stream>>>(x, W, bias, hT);
    gcn_aggregate_wmma<<<GN / 32, 256, 0, stream>>>(adj, hT, out);
}
